// FAdaLayer_nosequence_8177617732288
// MI455X (gfx1250) — matrix operand order repaired and checked by reading, not compiled, not run on hardware
//
#include <hip/hip_runtime.h>
#include <hip/hip_bf16.h>

#define HIDDEN   96
#define IN_DIM   256
#define NNODES   50000
#define NEDGES   800000

typedef _Float16 v16h __attribute__((ext_vector_type(16)));
typedef _Float16 v8h  __attribute__((ext_vector_type(8)));
typedef float    v8f  __attribute__((ext_vector_type(8)));

__device__ __forceinline__ v8f wmma_f16(v16h a, v16h b, v8f c) {
  // 8 args: (neg_a, A, neg_b, B, c_mod, C, reuse_a, reuse_b)
  return __builtin_amdgcn_wmma_f32_16x16x32_f16(false, a, false, b, (short)0, c,
                                                false, false);
}

__device__ __forceinline__ v8f zero_v8f() {
  v8f z;
#pragma unroll
  for (int i = 0; i < 8; ++i) z[i] = 0.0f;
  return z;
}

// A-fragment gather: row of f16 data, halves {k0+hi*16..+7, k0+hi*16+8..+7}
__device__ __forceinline__ v16h load_afrag(const _Float16* __restrict__ row,
                                           int k0, int hi) {
  const v8h lo8 = *(const v8h*)(row + k0 + hi * 16);
  const v8h up8 = *(const v8h*)(row + k0 + hi * 16 + 8);
  v16h a;
#pragma unroll
  for (int j = 0; j < 8; ++j) { a[j] = lo8[j]; a[8 + j] = up8[j]; }
  return a;
}

// ---------------------------------------------------------------- utilities
__global__ void zero_f32_kernel(float* __restrict__ p, int n) {
  int i = blockIdx.x * blockDim.x + threadIdx.x;
  if (i < n) p[i] = 0.0f;
}

__global__ void cvt_f32_to_f16_kernel(const float* __restrict__ in,
                                      _Float16* __restrict__ out, int n) {
  int i = blockIdx.x * blockDim.x + threadIdx.x;
  if (i < n) out[i] = (_Float16)in[i];
}

// ------------------------------------------------- h = x @ W_in^T + b_in
// One wave per 16-node tile. K=256 -> 8 k-steps of 32; N=96 -> 6 n-tiles.
__global__ __launch_bounds__(256)
void gemm_in_kernel(const float* __restrict__ x,
                    const _Float16* __restrict__ win16,
                    const float* __restrict__ bin,
                    float* __restrict__ h32,
                    _Float16* __restrict__ h16) {
  const int lane = threadIdx.x & 31;
  const int l  = lane & 15;
  const int hi = lane >> 4;
  const int tile = blockIdx.x * (blockDim.x >> 5) + (threadIdx.x >> 5);
  if (tile >= NNODES / 16) return;   // wave-uniform guard

  const int node_l = tile * 16 + l;
  const float* xrow = x + (long)node_l * IN_DIM;

  v8f acc[6];
#pragma unroll
  for (int nt = 0; nt < 6; ++nt) acc[nt] = zero_v8f();

#pragma unroll
  for (int ks = 0; ks < 8; ++ks) {
    const int k0 = ks * 32;
    v16h afrag;
#pragma unroll
    for (int j = 0; j < 8; ++j) {
      afrag[j]     = (_Float16)xrow[k0 + hi * 16 + j];
      afrag[8 + j] = (_Float16)xrow[k0 + hi * 16 + 8 + j];
    }
#pragma unroll
    for (int nt = 0; nt < 6; ++nt) {
      const v16h bfrag = *(const v16h*)(win16 +
          (long)(nt * 16 + l) * IN_DIM + k0 + hi * 16);
      acc[nt] = wmma_f16(afrag, bfrag, acc[nt]);
    }
  }

#pragma unroll
  for (int nt = 0; nt < 6; ++nt) {
    const int n = nt * 16 + l;
    const float bi = bin[n];
#pragma unroll
    for (int r = 0; r < 8; ++r) {
      const int node = tile * 16 + r + hi * 8;
      const float v = acc[nt][r] + bi;
      h32[(long)node * HIDDEN + n] = v;
      h16[(long)node * HIDDEN + n] = (_Float16)v;
    }
  }
}

// --------------------------------- edge gate GEMM + tanh + scatter message
// One wave per 32-edge tile-pair: two independent accumulator sets share
// every B fragment (2 WMMAs per dependent B load). K=192 as two K=96 passes:
//   part 0: h_dst x W_gate[:, 0:96], part 1: h_src x W_gate[:, 96:192]
__global__ __launch_bounds__(256)
void edge_kernel(const int* __restrict__ src, const int* __restrict__ dst,
                 const float* __restrict__ dvec,
                 const _Float16* __restrict__ h16,
                 const float* __restrict__ h32,
                 const _Float16* __restrict__ wg16,
                 const float* __restrict__ bgate,
                 float* __restrict__ aout,
                 float* __restrict__ zacc) {
  const int lane = threadIdx.x & 31;
  const int l  = lane & 15;
  const int hi = lane >> 4;
  const int pair = blockIdx.x * (blockDim.x >> 5) + (threadIdx.x >> 5);
  if (pair >= NEDGES / 32) return;   // wave-uniform guard

  const int base = pair * 32;        // tile0 = edges [base, base+16)
                                     // tile1 = edges [base+16, base+32)
  const int sIdxL0 = src[base + l];
  const int dIdxL0 = dst[base + l];
  const int sIdxL1 = src[base + 16 + l];
  const int dIdxL1 = dst[base + 16 + l];
  const float eL0 = dvec[sIdxL0] * dvec[dIdxL0];
  const float eL1 = dvec[sIdxL1] * dvec[dIdxL1];

  v8f acc0[6], acc1[6];
#pragma unroll
  for (int nt = 0; nt < 6; ++nt) { acc0[nt] = zero_v8f(); acc1[nt] = zero_v8f(); }

#pragma unroll
  for (int part = 0; part < 2; ++part) {
    const int row0 = (part == 0) ? dIdxL0 : sIdxL0;
    const int row1 = (part == 0) ? dIdxL1 : sIdxL1;
    const _Float16* arow0 = h16 + (long)row0 * HIDDEN;
    const _Float16* arow1 = h16 + (long)row1 * HIDDEN;
#pragma unroll
    for (int ks = 0; ks < 3; ++ks) {
      const int k0 = ks * 32;
      const v16h af0 = load_afrag(arow0, k0, hi);
      const v16h af1 = load_afrag(arow1, k0, hi);
#pragma unroll
      for (int nt = 0; nt < 6; ++nt) {
        // B fragment: contiguous 32B of f16 W_gate row (nt*16+l); shared by
        // both M-tiles -> 2 WMMAs amortize each dependent load.
        const v16h bfrag = *(const v16h*)(wg16 +
            (long)(nt * 16 + l) * (2 * HIDDEN) + part * HIDDEN + k0 + hi * 16);
        acc0[nt] = wmma_f16(af0, bfrag, acc0[nt]);
        acc1[nt] = wmma_f16(af1, bfrag, acc1[nt]);
      }
    }
  }

  // broadcast per-row (M = r+hi*8) edge metadata across the wave
  int sIdx0[8], dIdx0[8], sIdx1[8], dIdx1[8];
  float eM0[8], eM1[8];
#pragma unroll
  for (int r = 0; r < 8; ++r) {
    const int M = r + hi * 8;
    sIdx0[r] = __shfl(sIdxL0, M);
    dIdx0[r] = __shfl(dIdxL0, M);
    eM0[r]   = __shfl(eL0, M);
    sIdx1[r] = __shfl(sIdxL1, M);
    dIdx1[r] = __shfl(dIdxL1, M);
    eM1[r]   = __shfl(eL1, M);
  }

  const float inv_s = 0.07216878364870323f;  // 1/sqrt(2*HIDDEN)
#pragma unroll
  for (int nt = 0; nt < 6; ++nt) {
    const int n = nt * 16 + l;
    const float bg = bgate[n];
#pragma unroll
    for (int r = 0; r < 8; ++r) {
      const int M = r + hi * 8;
      // tile 0
      {
        const float av = tanhf((acc0[nt][r] + bg) * inv_s);
        const int edge = base + M;
        aout[(long)edge * HIDDEN + n] = av;
        const float hs = h32[(long)sIdx0[r] * HIDDEN + n];   // L2-resident
        atomicAdd(&zacc[(long)dIdx0[r] * HIDDEN + n], hs * av * eM0[r]);
      }
      // tile 1
      {
        const float av = tanhf((acc1[nt][r] + bg) * inv_s);
        const int edge = base + 16 + M;
        aout[(long)edge * HIDDEN + n] = av;
        const float hs = h32[(long)sIdx1[r] * HIDDEN + n];   // L2-resident
        atomicAdd(&zacc[(long)dIdx1[r] * HIDDEN + n], hs * av * eM1[r]);
      }
    }
  }
}

// ------------------------------------ z = log_softmax(zacc @ W_clf^T + b)
__global__ __launch_bounds__(256)
void clf_kernel(const float* __restrict__ zacc,
                const float* __restrict__ Wclf,
                const float* __restrict__ bclf,
                float* __restrict__ zout) {
  const int node = blockIdx.x * blockDim.x + threadIdx.x;
  if (node >= NNODES) return;
  const float* zr = zacc + (long)node * HIDDEN;
  float z0 = bclf[0], z1 = bclf[1];
#pragma unroll 4
  for (int k = 0; k < HIDDEN; ++k) {
    const float v = zr[k];
    z0 += v * Wclf[k];
    z1 += v * Wclf[HIDDEN + k];
  }
  const float m = fmaxf(z0, z1);
  const float lse = m + logf(expf(z0 - m) + expf(z1 - m));
  zout[node * 2 + 0] = z0 - lse;
  zout[node * 2 + 1] = z1 - lse;
}

// --------------------------------------------------------------- launcher
extern "C" void kernel_launch(void* const* d_in, const int* in_sizes, int n_in,
                              void* d_out, int out_size, void* d_ws, size_t ws_size,
                              hipStream_t stream) {
  const float* x    = (const float*)d_in[0];
  const float* dvec = (const float*)d_in[1];
  const int*   src  = (const int*)d_in[2];
  const int*   dst  = (const int*)d_in[3];
  const float* Win  = (const float*)d_in[4];
  const float* bin  = (const float*)d_in[5];
  const float* Wg   = (const float*)d_in[6];
  const float* bg   = (const float*)d_in[7];
  const float* Wclf = (const float*)d_in[8];
  const float* bclf = (const float*)d_in[9];

  // workspace layout (all offsets 256B-aligned)
  char* ws = (char*)d_ws;
  float*    h32   = (float*)(ws + 0);            // 50000*96*4 = 19,200,000 B
  _Float16* h16   = (_Float16*)(ws + 19200000);  //  9,600,000 B
  _Float16* wg16  = (_Float16*)(ws + 28800000);  //     36,864 B
  _Float16* win16 = (_Float16*)(ws + 28836864);  //     49,152 B
  float*    zacc  = (float*)(ws + 28886016);     // 19,200,000 B (total ~48.1 MB)

  float* zout = (float*)d_out;            // [50000, 2]
  float* aout = (float*)d_out + 100000;   // [800000, 96]

  // 1) zero the scatter accumulator (must be per-launch for determinism)
  {
    const int n = NNODES * HIDDEN;
    zero_f32_kernel<<<(n + 255) / 256, 256, 0, stream>>>(zacc, n);
  }
  // 2) weights -> f16 (W_gate and W_in)
  {
    const int ng = HIDDEN * 2 * HIDDEN;            // 18432
    cvt_f32_to_f16_kernel<<<(ng + 255) / 256, 256, 0, stream>>>(Wg, wg16, ng);
    const int nw = HIDDEN * IN_DIM;                // 24576
    cvt_f32_to_f16_kernel<<<(nw + 255) / 256, 256, 0, stream>>>(Win, win16, nw);
  }
  // 3) input GEMM: 3125 tiles of 16 nodes, 8 waves/block
  {
    const int tiles = NNODES / 16;                 // 3125 exactly
    const int blocks = (tiles + 7) / 8;
    gemm_in_kernel<<<blocks, 256, 0, stream>>>(x, win16, bin, h32, h16);
  }
  // 4) edge GEMM + gate + scatter: 25000 tile-pairs of 32 edges, 8 waves/block
  {
    const int pairs = NEDGES / 32;                 // 25000 exactly
    const int blocks = (pairs + 7) / 8;            // 3125
    edge_kernel<<<blocks, 256, 0, stream>>>(src, dst, dvec, h16, h32, wg16, bg,
                                            aout, zacc);
  }
  // 5) classifier + log_softmax
  {
    clf_kernel<<<(NNODES + 255) / 256, 256, 0, stream>>>(zacc, Wclf, bclf, zout);
  }
}
